// VitDetAttention_17746804867918
// MI455X (gfx1250) — matrix operand order repaired and checked by reading, not compiled, not run on hardware
//
#include <hip/hip_runtime.h>

// ---------------------------------------------------------------------------
// ViTDet attention for MI455X (gfx1250, wave32, WMMA bf16 16x16x32, f32 acc)
// Attention K/V streamed through LDS with async-to-LDS double buffering.
// ---------------------------------------------------------------------------

typedef __bf16 bf16_t;
typedef __attribute__((ext_vector_type(16))) __bf16 v16bf;
typedef __attribute__((ext_vector_type(8)))  float  v8f;

union Frag { v16bf v; uint4 q[2]; };

__device__ __forceinline__ unsigned short f32_to_bf16_bits(float f) {
  unsigned u = __builtin_bit_cast(unsigned, f);
  u += 0x7FFFu + ((u >> 16) & 1u);          // round-to-nearest-even
  return (unsigned short)(u >> 16);
}
__device__ __forceinline__ bf16_t to_bf16(float f) {
  return __builtin_bit_cast(bf16_t, f32_to_bf16_bits(f));
}
__device__ __forceinline__ float bf16_to_f32(bf16_t b) {
  unsigned u = ((unsigned)__builtin_bit_cast(unsigned short, b)) << 16;
  return __builtin_bit_cast(float, u);
}
__device__ __forceinline__ uint4 ld_u4(const bf16_t* p) {
  return *reinterpret_cast<const uint4*>(p);
}
__device__ __forceinline__ v8f wmma_bf16(const Frag& a, const Frag& b, v8f c) {
  return __builtin_amdgcn_wmma_f32_16x16x32_bf16(false, a.v, false, b.v,
                                                 (short)0, c, false, false);
}
// CDNA5 async copy: 16 bytes/lane, global -> LDS, tracked by ASYNCcnt.
__device__ __forceinline__ void async_cp16(unsigned lds_off, const bf16_t* g) {
  unsigned long long ga = (unsigned long long)(uintptr_t)g;
  asm volatile("global_load_async_to_lds_b128 %0, %1, off"
               :: "v"(lds_off), "v"(ga) : "memory");
}
__device__ __forceinline__ void wait_async0() {
  asm volatile("s_wait_asynccnt 0x0" ::: "memory");
}

// Problem constants
#define BB   8
#define CC   768
#define NH   12
#define HD   64
#define HWN  1024            // H*W
#define BHN  96              // B*NH
#define MM   8192            // B*H*W tokens
#define N3   2304            // 3*C

// Workspace layout (bytes)
static constexpr size_t OFF_XB    = 0;
static constexpr size_t SZ_XB     = (size_t)MM * CC * 2;
static constexpr size_t OFF_WQKV  = OFF_XB + SZ_XB;
static constexpr size_t SZ_WQKV   = (size_t)CC * N3 * 2;
static constexpr size_t OFF_WPROJ = OFF_WQKV + SZ_WQKV;
static constexpr size_t SZ_WPROJ  = (size_t)CC * CC * 2;
static constexpr size_t OFF_QS    = OFF_WPROJ + SZ_WPROJ;
static constexpr size_t SZ_QS     = (size_t)BHN * HWN * HD * 2;
static constexpr size_t OFF_KT    = OFF_QS + SZ_QS;
static constexpr size_t OFF_V     = OFF_KT + SZ_QS;
static constexpr size_t OFF_RELH  = OFF_V + SZ_QS;
static constexpr size_t SZ_REL    = (size_t)BHN * HWN * 32 * 4;
static constexpr size_t OFF_RELW  = OFF_RELH + SZ_REL;
static constexpr size_t OFF_OB    = OFF_RELW + SZ_REL;

// ---------------------------------------------------------------------------
// 0) f32 -> bf16 conversion (vectorized x4)
// ---------------------------------------------------------------------------
__global__ void __launch_bounds__(256)
cvt_f32_bf16(const float* __restrict__ src, bf16_t* __restrict__ dst, int n4) {
  int i = blockIdx.x * 256 + threadIdx.x;
  if (i >= n4) return;
  const float4 f = reinterpret_cast<const float4*>(src)[i];
  uint2 p;
  p.x = (unsigned)f32_to_bf16_bits(f.x) | ((unsigned)f32_to_bf16_bits(f.y) << 16);
  p.y = (unsigned)f32_to_bf16_bits(f.z) | ((unsigned)f32_to_bf16_bits(f.w) << 16);
  reinterpret_cast<uint2*>(dst)[i] = p;
}

// ---------------------------------------------------------------------------
// 1) QKV GEMM: [8192,768] x [768,2304] + bias; epilogue scatters Q*scale,
//    K transposed (d-major), V into per-head bf16 buffers.
// ---------------------------------------------------------------------------
__global__ void __launch_bounds__(256)
qkv_gemm(const bf16_t* __restrict__ xb, const bf16_t* __restrict__ wb,
         const float* __restrict__ bqkv,
         bf16_t* __restrict__ Qs, bf16_t* __restrict__ Kt, bf16_t* __restrict__ Vv) {
  const unsigned lane = threadIdx.x & 31u;
  const unsigned wv   = threadIdx.x >> 5;
  const unsigned n0   = blockIdx.x * 16;
  const unsigned m0   = blockIdx.y * 512 + wv * 64;
  const unsigned hlf  = lane >> 4;
  const unsigned l16  = lane & 15u;
  const unsigned kbo  = hlf * 8;

  v8f acc[4] = {};
  for (unsigned k0 = 0; k0 < CC; k0 += 32) {
    Frag bfr;                              // B: lane = N column, 16 halfs = K (table order)
    const bf16_t* wrow = wb + (size_t)(k0 + kbo) * N3 + n0 + l16;
    for (int s = 0; s < 16; ++s) bfr.v[s] = wrow[(size_t)((s & 7) + (s >> 3) * 16) * N3];
#pragma unroll
    for (int t = 0; t < 4; ++t) {
      Frag af;
      const bf16_t* arow = xb + (size_t)(m0 + t * 16 + l16) * CC + k0 + kbo;
      af.q[0] = ld_u4(arow); af.q[1] = ld_u4(arow + 16);
      acc[t] = wmma_bf16(af, bfr, acc[t]);
    }
  }

  const unsigned n     = n0 + l16;
  const float    bias  = bqkv[n];
  const unsigned which = n / CC;
  const unsigned r     = n % CC;
  const unsigned head  = r >> 6;
  const unsigned d     = r & 63u;
#pragma unroll
  for (int t = 0; t < 4; ++t) {
#pragma unroll
    for (int i = 0; i < 8; ++i) {
      const unsigned m  = m0 + t * 16 + i + hlf * 8;
      const unsigned b  = m >> 10;
      const unsigned hw = m & 1023u;
      const unsigned bh = b * NH + head;
      const float val = acc[t][i] + bias;
      if (which == 0)
        Qs[((size_t)bh * HWN + hw) * HD + d] = to_bf16(val * 0.125f); // hd^-0.5
      else if (which == 1)
        Kt[((size_t)bh * HD + d) * HWN + hw] = to_bf16(val);
      else
        Vv[((size_t)bh * HWN + hw) * HD + d] = to_bf16(val);
    }
  }
}

// ---------------------------------------------------------------------------
// 2) Decomposed rel-pos bias tables: relH/relW [96,1024,32]
// ---------------------------------------------------------------------------
__global__ void __launch_bounds__(256)
rel_bias(const bf16_t* __restrict__ Qs, const float* __restrict__ rph,
         const float* __restrict__ rpw, float* __restrict__ relH,
         float* __restrict__ relW) {
  const unsigned idx = blockIdx.x * 256 + threadIdx.x;   // 96*1024*64 exact
  const int kk = (int)(idx & 63u);
  const int q  = (int)((idx >> 6) & 1023u);
  const int bh = (int)(idx >> 16);
  const int hq = q >> 5, wq = q & 31;
  const bf16_t* qrow = Qs + ((size_t)bh * HWN + q) * HD;
  const float* rrow;
  float* out;
  if (kk < 32) {
    rrow = rph + (size_t)(hq - kk + 31) * HD;
    out  = relH + ((size_t)bh * HWN + q) * 32 + kk;
  } else {
    const int wk = kk - 32;
    rrow = rpw + (size_t)(wq - wk + 31) * HD;
    out  = relW + ((size_t)bh * HWN + q) * 32 + wk;
  }
  float s = 0.f;
#pragma unroll 8
  for (int d = 0; d < HD; ++d) s += bf16_to_f32(qrow[d]) * rrow[d];
  *out = s * 8.0f;   // undo the 0.125 pre-scale baked into Qs
}

// ---------------------------------------------------------------------------
// 3) Flash attention. Block = 8 waves x 16 queries = 128 queries.
//    K/V streamed in 64-key chunks through LDS via async-to-LDS copies,
//    double-buffered so DMA of chunk c+1 overlaps compute on chunk c.
//    Per 32-key step: 4 WMMA (S=QK^T) + bias + online softmax + 4 WMMA (PV).
// ---------------------------------------------------------------------------
__global__ void __launch_bounds__(256)
attn_kernel(const bf16_t* __restrict__ Qs, const bf16_t* __restrict__ Kt,
            const bf16_t* __restrict__ Vv, const float* __restrict__ relH,
            const float* __restrict__ relW, bf16_t* __restrict__ O) {
  __shared__ bf16_t ksh[2][64][72];        // [buf][d][key]   (+8 pad, 16B rows)
  __shared__ bf16_t vsh[2][64][72];        // [buf][key][d]
  __shared__ bf16_t pbuf[8][16][48];       // per-wave P re-layout staging

  const unsigned tid  = threadIdx.x;
  const unsigned lane = tid & 31u;
  const unsigned wv   = tid >> 5;
  const unsigned bh   = blockIdx.x;
  const unsigned q0   = blockIdx.y * 128 + wv * 16;
  const unsigned hlf  = lane >> 4;
  const unsigned l16  = lane & 15u;
  const unsigned b    = bh / NH;
  const unsigned head = bh % NH;

  // cooperative async staging of one 64-key chunk of K^T and V into LDS
  auto stage_chunk = [&](unsigned c, unsigned dstbuf) {
#pragma unroll
    for (int rep = 0; rep < 2; ++rep) {
      const unsigned idx = tid + (unsigned)rep * 256u;   // 0..511
      const unsigned row = idx >> 3;                     // 0..63
      const unsigned seg = (idx & 7u) * 8u;              // half-word offset
      const bf16_t* kg = Kt + ((size_t)bh * HD + row) * HWN + c * 64 + seg;
      async_cp16((unsigned)(uintptr_t)&ksh[dstbuf][row][seg], kg);
      const bf16_t* vg = Vv + ((size_t)bh * HWN + c * 64 + row) * HD + seg;
      async_cp16((unsigned)(uintptr_t)&vsh[dstbuf][row][seg], vg);
    }
  };

  // Q A-fragments (K-dim = head dim 64, two 32-chunks), registers for life
  Frag a0, a1;
  {
    const bf16_t* qrow = Qs + ((size_t)bh * HWN + q0 + l16) * HD + hlf * 8;
    a0.q[0] = ld_u4(qrow);      a0.q[1] = ld_u4(qrow + 16);
    a1.q[0] = ld_u4(qrow + 32); a1.q[1] = ld_u4(qrow + 48);
  }

  v8f o0 = {}, o1 = {}, o2 = {}, o3 = {};
  float mrow[8], lrow[8];
#pragma unroll
  for (int i = 0; i < 8; ++i) { mrow[i] = -3.0e38f; lrow[i] = 0.f; }

  const float* rHrow = relH + (size_t)bh * HWN * 32;
  const float* rWrow = relW + (size_t)bh * HWN * 32;

  stage_chunk(0, 0);
  wait_async0();
  __syncthreads();

  for (unsigned c = 0; c < 16; ++c) {
    const unsigned cur = c & 1u;
    if (c + 1 < 16) stage_chunk(c + 1, cur ^ 1u);   // overlap DMA with compute

#pragma unroll
    for (unsigned sub = 0; sub < 64; sub += 32) {
      const unsigned k0 = c * 64 + sub;

      // K^T B-fragments from LDS: lane = key column, 16 halfs = d (table order)
      Frag kb00, kb01, kb10, kb11;
      {
        for (int s = 0; s < 16; ++s) {
          const unsigned dd = hlf * 8 + (s & 7) + (s >> 3) * 16;
          kb00.v[s] = ksh[cur][dd][sub + l16];
          kb01.v[s] = ksh[cur][dd][sub + 16 + l16];
          kb10.v[s] = ksh[cur][dd + 32][sub + l16];
          kb11.v[s] = ksh[cur][dd + 32][sub + 16 + l16];
        }
      }

      v8f s_lo = {}, s_hi = {};
      s_lo = wmma_bf16(a0, kb00, s_lo); s_lo = wmma_bf16(a1, kb10, s_lo);
      s_hi = wmma_bf16(a0, kb01, s_hi); s_hi = wmma_bf16(a1, kb11, s_hi);

      // bias + online softmax (rows live across 16-lane halves)
      const unsigned keyL = k0 + l16, keyH = keyL + 16;
      const int hkL = (int)(keyL >> 5), wkL = (int)(keyL & 31u);
      const int hkH = (int)(keyH >> 5), wkH = (int)(keyH & 31u);
      float pl[8], ph[8];
#pragma unroll
      for (int i = 0; i < 8; ++i) {
        const unsigned qq = q0 + i + hlf * 8;
        const float bL = rHrow[qq * 32 + hkL] + rWrow[qq * 32 + wkL];
        const float bH = rHrow[qq * 32 + hkH] + rWrow[qq * 32 + wkH];
        const float sl = s_lo[i] + bL;
        const float sh = s_hi[i] + bH;
        float mx = fmaxf(sl, sh);
        mx = fmaxf(mx, __shfl_xor(mx, 1, 32));
        mx = fmaxf(mx, __shfl_xor(mx, 2, 32));
        mx = fmaxf(mx, __shfl_xor(mx, 4, 32));
        mx = fmaxf(mx, __shfl_xor(mx, 8, 32));
        const float mnew = fmaxf(mrow[i], mx);
        const float corr = __expf(mrow[i] - mnew);
        mrow[i] = mnew;
        const float el = __expf(sl - mnew);
        const float eh = __expf(sh - mnew);
        float rs = el + eh;
        rs += __shfl_xor(rs, 1, 32);
        rs += __shfl_xor(rs, 2, 32);
        rs += __shfl_xor(rs, 4, 32);
        rs += __shfl_xor(rs, 8, 32);
        lrow[i] = lrow[i] * corr + rs;
        o0[i] *= corr; o1[i] *= corr; o2[i] *= corr; o3[i] *= corr;
        pl[i] = el; ph[i] = eh;
      }

      // stage P (C-layout) into LDS, read back as A-fragment (16x32, K=keys)
#pragma unroll
      for (int i = 0; i < 8; ++i) {
        const unsigned r = i + hlf * 8;
        pbuf[wv][r][l16]      = to_bf16(pl[i]);
        pbuf[wv][r][16 + l16] = to_bf16(ph[i]);
      }
      Frag pa;
      {
        const bf16_t* prow = &pbuf[wv][l16][hlf * 8];
        pa.q[0] = ld_u4(prow);
        pa.q[1] = ld_u4(prow + 16);
      }

      // V B-fragments from LDS: lane = d column, 16 halfs = keys (table order)
      const bf16_t* vrow = &vsh[cur][sub + hlf * 8][l16];
      Frag vb;
      for (int s = 0; s < 16; ++s) vb.v[s] = vrow[((s & 7) + (s >> 3) * 16) * 72];      o0 = wmma_bf16(pa, vb, o0);
      for (int s = 0; s < 16; ++s) vb.v[s] = vrow[((s & 7) + (s >> 3) * 16) * 72 + 16]; o1 = wmma_bf16(pa, vb, o1);
      for (int s = 0; s < 16; ++s) vb.v[s] = vrow[((s & 7) + (s >> 3) * 16) * 72 + 32]; o2 = wmma_bf16(pa, vb, o2);
      for (int s = 0; s < 16; ++s) vb.v[s] = vrow[((s & 7) + (s >> 3) * 16) * 72 + 48]; o3 = wmma_bf16(pa, vb, o3);
    }

    wait_async0();        // our wave's DMA for the next chunk has landed
    __syncthreads();      // everyone's has; safe to flip buffers
  }

  // normalize + store head-interleaved [B, HW, nh*hd]
#pragma unroll
  for (int i = 0; i < 8; ++i) {
    const float inv = 1.0f / lrow[i];
    const unsigned qq = q0 + i + hlf * 8;
    bf16_t* orow = O + ((size_t)b * HWN + qq) * CC + head * HD;
    orow[0  + l16] = to_bf16(o0[i] * inv);
    orow[16 + l16] = to_bf16(o1[i] * inv);
    orow[32 + l16] = to_bf16(o2[i] * inv);
    orow[48 + l16] = to_bf16(o3[i] * inv);
  }
}

// ---------------------------------------------------------------------------
// 4) Output projection: out = O @ w_proj + b_proj -> f32 d_out
// ---------------------------------------------------------------------------
__global__ void __launch_bounds__(256)
proj_gemm(const bf16_t* __restrict__ Ob, const bf16_t* __restrict__ wb,
          const float* __restrict__ bias, float* __restrict__ out) {
  const unsigned lane = threadIdx.x & 31u;
  const unsigned wv   = threadIdx.x >> 5;
  const unsigned n0   = blockIdx.x * 16;
  const unsigned m0   = blockIdx.y * 512 + wv * 64;
  const unsigned hlf  = lane >> 4;
  const unsigned l16  = lane & 15u;
  const unsigned kbo  = hlf * 8;

  v8f acc[4] = {};
  for (unsigned k0 = 0; k0 < CC; k0 += 32) {
    Frag bfr;
    const bf16_t* wrow = wb + (size_t)(k0 + kbo) * CC + n0 + l16;
    for (int s = 0; s < 16; ++s) bfr.v[s] = wrow[(size_t)((s & 7) + (s >> 3) * 16) * CC];
#pragma unroll
    for (int t = 0; t < 4; ++t) {
      Frag af;
      const bf16_t* arow = Ob + (size_t)(m0 + t * 16 + l16) * CC + k0 + kbo;
      af.q[0] = ld_u4(arow); af.q[1] = ld_u4(arow + 16);
      acc[t] = wmma_bf16(af, bfr, acc[t]);
    }
  }
  const unsigned n = n0 + l16;
  const float bn = bias[n];
#pragma unroll
  for (int t = 0; t < 4; ++t) {
#pragma unroll
    for (int i = 0; i < 8; ++i) {
      const unsigned m = m0 + t * 16 + i + hlf * 8;
      out[(size_t)m * CC + n] = acc[t][i] + bn;
    }
  }
}

// ---------------------------------------------------------------------------
extern "C" void kernel_launch(void* const* d_in, const int* in_sizes, int n_in,
                              void* d_out, int out_size, void* d_ws, size_t ws_size,
                              hipStream_t stream) {
  (void)in_sizes; (void)n_in; (void)out_size; (void)ws_size;
  const float* x      = (const float*)d_in[0];
  const float* w_qkv  = (const float*)d_in[1];
  const float* b_qkv  = (const float*)d_in[2];
  const float* w_proj = (const float*)d_in[3];
  const float* b_proj = (const float*)d_in[4];
  const float* rph    = (const float*)d_in[5];
  const float* rpw    = (const float*)d_in[6];
  float* out = (float*)d_out;

  char* ws = (char*)d_ws;
  bf16_t* xb     = (bf16_t*)(ws + OFF_XB);
  bf16_t* wqkvb  = (bf16_t*)(ws + OFF_WQKV);
  bf16_t* wprojb = (bf16_t*)(ws + OFF_WPROJ);
  bf16_t* Qs     = (bf16_t*)(ws + OFF_QS);
  bf16_t* Kt     = (bf16_t*)(ws + OFF_KT);
  bf16_t* Vv     = (bf16_t*)(ws + OFF_V);
  float*  relH   = (float*)(ws + OFF_RELH);
  float*  relW   = (float*)(ws + OFF_RELW);
  bf16_t* Ob     = (bf16_t*)(ws + OFF_OB);

  {
    int n4 = (MM * CC) / 4;
    cvt_f32_bf16<<<dim3((n4 + 255) / 256), dim3(256), 0, stream>>>(x, xb, n4);
    n4 = (CC * N3) / 4;
    cvt_f32_bf16<<<dim3((n4 + 255) / 256), dim3(256), 0, stream>>>(w_qkv, wqkvb, n4);
    n4 = (CC * CC) / 4;
    cvt_f32_bf16<<<dim3((n4 + 255) / 256), dim3(256), 0, stream>>>(w_proj, wprojb, n4);
  }
  qkv_gemm<<<dim3(N3 / 16, MM / 512), dim3(256), 0, stream>>>(xb, wqkvb, b_qkv,
                                                              Qs, Kt, Vv);
  rel_bias<<<dim3((BHN * HWN * 64) / 256), dim3(256), 0, stream>>>(Qs, rph, rpw,
                                                                   relH, relW);
  attn_kernel<<<dim3(BHN, HWN / 128), dim3(256), 0, stream>>>(Qs, Kt, Vv, relH,
                                                              relW, Ob);
  proj_gemm<<<dim3(CC / 16, MM / 512), dim3(256), 0, stream>>>(Ob, wprojb,
                                                               b_proj, out);
}
